// SpatialTransformer_26749056319808
// MI455X (gfx1250) — matrix operand order repaired and checked by reading, not compiled, not run on hardware
//
#include <hip/hip_runtime.h>
#include <cmath>

// ---------------------------------------------------------------------------
// Types for CDNA5 WMMA (wave32): V_WMMA_F32_16X16X32_F16
// ---------------------------------------------------------------------------
typedef __attribute__((ext_vector_type(16))) _Float16 v16h;
typedef __attribute__((ext_vector_type(8)))  _Float16 v8h;
typedef __attribute__((ext_vector_type(8)))  float    v8f;

__device__ __forceinline__ v16h make_v16(v8h lo, v8h hi) {
    return __builtin_shufflevector(lo, hi, 0, 1, 2, 3, 4, 5, 6, 7,
                                           8, 9, 10, 11, 12, 13, 14, 15);
}

// ---- Fragment loaders per CDNA5 ISA §7.12.2 (vectorized, ds_load_b128) -----
// A-matrix 16x32 f16 from ROW-major src[m][k]: lane L holds row m = L&15;
// halves k = 8*hl..8*hl+7 and 16+8*hl..+7 (hl = L>>4) -> two contiguous v8h.
__device__ __forceinline__ v16h load_frag_a(const _Float16* src, int ld, int lane) {
    int m  = lane & 15;
    int hl = lane >> 4;
    const _Float16* p = src + m * ld + 8 * hl;
    v8h lo = *(const v8h*)p;
    v8h hi = *(const v8h*)(p + 16);
    return make_v16(lo, hi);
}

// B-matrix 32x16 f16 where logical B[k][n] is stored COLUMN-major:
// value(k,n) = srcT[n*ld + k]. Lane L holds col n = L&15; halves
// k = 8*hl..8*hl+7 and 16+8*hl..+7 -> two contiguous v8h.
__device__ __forceinline__ v16h load_frag_b_t(const _Float16* srcT, int ld, int lane) {
    int n  = lane & 15;
    int hl = lane >> 4;
    const _Float16* p = srcT + n * ld + 8 * hl;
    v8h lo = *(const v8h*)p;
    v8h hi = *(const v8h*)(p + 16);
    return make_v16(lo, hi);
}

// ---------------------------------------------------------------------------
// Elementwise helpers
// ---------------------------------------------------------------------------
__global__ void k_f32_to_f16(const float* __restrict__ src, _Float16* __restrict__ dst, int n) {
    int i = blockIdx.x * 256 + threadIdx.x;
    if (i < n) dst[i] = (_Float16)src[i];
}

__global__ void k_fill0_f16(_Float16* __restrict__ dst, int n) {
    int i = blockIdx.x * 256 + threadIdx.x;
    if (i < n) dst[i] = (_Float16)0.f;
}

// ---------------------------------------------------------------------------
// GroupNorm over x[1,320,64,64], 32 groups of 10 channels.
// Writes normalized activations TRANSPOSED to [s=4096][c=320] f16.
// ---------------------------------------------------------------------------
__global__ void k_groupnorm(const float* __restrict__ x,
                            const float* __restrict__ gs,
                            const float* __restrict__ gb,
                            _Float16* __restrict__ out) {
    __shared__ float r1[256];
    __shared__ float r2[256];
    const int CPG = 10, S = 4096, N = CPG * S;
    int g = blockIdx.x, tid = threadIdx.x;
    int c0 = g * CPG;
    float s = 0.f, s2 = 0.f;
    for (int i = tid; i < N; i += 256) {
        float v = x[(size_t)(c0 + i / S) * S + (i % S)];
        s += v; s2 += v * v;
    }
    r1[tid] = s; r2[tid] = s2;
    __syncthreads();
    for (int st = 128; st > 0; st >>= 1) {
        if (tid < st) { r1[tid] += r1[tid + st]; r2[tid] += r2[tid + st]; }
        __syncthreads();
    }
    float mu  = r1[0] / (float)N;
    float var = r2[0] / (float)N - mu * mu;
    float rs  = rsqrtf(var + 1e-6f);
    for (int i = tid; i < N; i += 256) {
        int c = c0 + i / S, sp = i % S;
        float v = (x[(size_t)c * S + sp] - mu) * rs * gs[c] + gb[c];
        out[(size_t)sp * 320 + c] = (_Float16)v;
    }
}

// ---------------------------------------------------------------------------
// LayerNorm over rows of h[4096,320] (f32), writes f16 [4096,320].
// ---------------------------------------------------------------------------
__global__ void k_layernorm(const float* __restrict__ h,
                            const float* __restrict__ sc,
                            const float* __restrict__ bi,
                            _Float16* __restrict__ out) {
    __shared__ float r1[128];
    __shared__ float r2[128];
    const int D = 320;
    int r = blockIdx.x, tid = threadIdx.x;
    const float* row = h + (size_t)r * D;
    float s = 0.f, s2 = 0.f;
    for (int i = tid; i < D; i += 128) { float v = row[i]; s += v; s2 += v * v; }
    r1[tid] = s; r2[tid] = s2;
    __syncthreads();
    for (int st = 64; st > 0; st >>= 1) {
        if (tid < st) { r1[tid] += r1[tid + st]; r2[tid] += r2[tid + st]; }
        __syncthreads();
    }
    float mu  = r1[0] / (float)D;
    float var = r2[0] / (float)D - mu * mu;
    float rs  = rsqrtf(var + 1e-5f);
    for (int i = tid; i < D; i += 128)
        out[(size_t)r * D + i] = (_Float16)((row[i] - mu) * rs * sc[i] + bi[i]);
}

// ---------------------------------------------------------------------------
// GEGLU: y[4096,2560] f16 -> out[4096,1280] f16 = a * gelu_exact(g)
// ---------------------------------------------------------------------------
__global__ void k_geglu(const _Float16* __restrict__ y, _Float16* __restrict__ out, int M) {
    int idx = blockIdx.x * 256 + threadIdx.x;
    int total = M * 1280;
    if (idx >= total) return;
    int m = idx / 1280, j = idx - m * 1280;
    float a = (float)y[(size_t)m * 2560 + j];
    float g = (float)y[(size_t)m * 2560 + 1280 + j];
    float gel = 0.5f * g * (1.0f + erff(g * 0.7071067811865475f));
    out[idx] = (_Float16)(a * gel);
}

// ---------------------------------------------------------------------------
// Generic tiled WMMA GEMM: C(MxN) = A(MxK,f16) * B(KxN,f16) [+bias][+res]
// Block = 128 threads (4 waves), 64x64 output tile, BK=32.
// A staged row-major (vector b128 in/out); B staged COLUMN-major so fragment
// reads are contiguous b128.
// Epilogue modes:
//   0: f32 store C[m*ldc+n] (+res same layout)
//   1: f16 store C[m*ldc+n]
//   2: f16 scatter into head-padded QKV [head][m][64], head=n/40, d=n%40
//   3: f32 TRANSPOSED store C[n*ldc+m] (+res same transposed layout)
// ---------------------------------------------------------------------------
struct GemmP {
    const _Float16* A;
    const _Float16* B;
    const float* bias;   // nullable
    const float* res;    // nullable
    float*    outf;      // modes 0,3
    _Float16* outh;      // modes 1,2
    int M, N, K, lda, ldb, ldc, mode, qkv_mstride;
};

__global__ __launch_bounds__(128) void k_wmma_gemm(GemmP p) {
    __shared__ _Float16 Asm[64][40];    // row-major [m][k], +8 pad
    __shared__ _Float16 BsmT[64][40];   // column-major [n][k], +8 pad
    int tid  = threadIdx.x;
    int lane = tid & 31;
    int w    = tid >> 5;
    int bn = blockIdx.x * 64;
    int bm = blockIdx.y * 64;

    v8f acc[4] = {};
    int ksteps = (p.K + 31) >> 5;

    for (int kt = 0; kt < ksteps; ++kt) {
        int k0 = kt << 5;
        __syncthreads();
        // Stage A tile 64x32: 256 v8h chunks, global b128 -> ds b128
        for (int ch = tid; ch < 256; ch += 128) {
            int r = ch >> 2, co = (ch & 3) * 8;
            int gm = bm + r;
            v8h v = {};
            if (gm < p.M && k0 + co + 8 <= p.K)
                v = *(const v8h*)(p.A + (size_t)gm * p.lda + k0 + co);
            *(v8h*)&Asm[r][co] = v;
        }
        // Stage B tile 32x64 transposed: global b128 row read, 8x b16 scatter
        for (int ch = tid; ch < 256; ch += 128) {
            int kr = ch >> 3, no = (ch & 7) * 8;
            int gk = k0 + kr;
            v8h v = {};
            if (gk < p.K && bn + no + 8 <= p.N)
                v = *(const v8h*)(p.B + (size_t)gk * p.ldb + bn + no);
#pragma unroll
            for (int e = 0; e < 8; ++e) BsmT[no + e][kr] = v[e];
        }
        // Prefetch next k-tile (global_prefetch_b8 on gfx1250)
        if (kt + 1 < ksteps) {
            int gmP = bm + (tid >> 1);
            if (gmP < p.M) __builtin_prefetch(p.A + (size_t)gmP * p.lda + k0 + 32, 0, 0);
            __builtin_prefetch(p.B + (size_t)(k0 + 32 + (tid & 31)) * p.ldb + bn, 0, 0);
        }
        __syncthreads();

        v16h a = load_frag_a(&Asm[w * 16][0], 40, lane);
#pragma unroll
        for (int nt = 0; nt < 4; ++nt) {
            v16h b = load_frag_b_t(&BsmT[nt * 16][0], 40, lane);
            acc[nt] = __builtin_amdgcn_wmma_f32_16x16x32_f16(
                false, a, false, b, (short)0, acc[nt], false, false);
        }
    }

    // Epilogue
    int n  = lane & 15;
    int mb = (lane >> 4) * 8;
#pragma unroll
    for (int nt = 0; nt < 4; ++nt) {
        int gn = bn + nt * 16 + n;
#pragma unroll
        for (int r = 0; r < 8; ++r) {
            int gm = bm + w * 16 + mb + r;
            if (gm >= p.M || gn >= p.N) continue;
            float v = acc[nt][r];
            if (p.bias) v += p.bias[gn];
            if (p.mode == 0) {
                size_t o = (size_t)gm * p.ldc + gn;
                if (p.res) v += p.res[o];
                p.outf[o] = v;
            } else if (p.mode == 1) {
                p.outh[(size_t)gm * p.ldc + gn] = (_Float16)v;
            } else if (p.mode == 2) {
                int hd = gn / 40, dd = gn - hd * 40;
                p.outh[((size_t)hd * p.qkv_mstride + gm) * 64 + dd] = (_Float16)v;
            } else { // 3: transposed f32
                size_t o = (size_t)gn * p.ldc + gm;
                float rv = p.res ? p.res[o] : 0.f;
                p.outf[o] = v + rv;
            }
        }
    }
}

// ---------------------------------------------------------------------------
// Flash attention: block = (head, 64-query tile), 128 threads = 4 waves,
// each wave owns 16 query rows. Head dim padded 40 -> 64 (zeros).
// Q/K/V: f16 [head][tokens][64]. O: f16 [Sq][320] (n = head*40+d).
//
// Online softmax fully in registers: S-tile C-fragment rows (m = r + 8*hl)
// match the O-accumulator rows exactly, so running max/sum/alpha live in
// per-lane arrays; row reductions are 4-step __shfl_xor within 16-lane halves.
// Only P transits LDS (C-layout -> A-layout transpose). V stored transposed
// in LDS so the P*V B-fragments are contiguous b128 reads.
// ---------------------------------------------------------------------------
__global__ __launch_bounds__(128) void k_flash_attn(
    const _Float16* __restrict__ Q, const _Float16* __restrict__ K,
    const _Float16* __restrict__ V, _Float16* __restrict__ O,
    int Sq, int Tvalid, int Tpad, float scale) {
    __shared__ _Float16 Qs[64][72];       // [q][d]   row-major
    __shared__ _Float16 Ks[64][72];       // [key][d] row-major (read as B^T)
    __shared__ _Float16 Vt[64][72];       // [d][key] transposed
    __shared__ _Float16 Psm[4][16][72];   // per-wave [q][key]

    int tid  = threadIdx.x;
    int lane = tid & 31;
    int w    = tid >> 5;
    int qbase = blockIdx.x * 64;
    int head  = blockIdx.y;
    int n  = lane & 15;
    int mb = (lane >> 4) * 8;

    const _Float16* Qg = Q + ((size_t)head * Sq + qbase) * 64;
    for (int ch = tid; ch < 512; ch += 128) {
        int r = ch >> 3, co = (ch & 7) * 8;
        *(v8h*)&Qs[r][co] = *(const v8h*)(Qg + (size_t)r * 64 + co);
    }

    float mrow[8], lrow[8];
#pragma unroll
    for (int r = 0; r < 8; ++r) { mrow[r] = -INFINITY; lrow[r] = 0.f; }
    v8f acc[4] = {};

    int nchunk = Tpad >> 6;
    for (int ch = 0; ch < nchunk; ++ch) {
        int kb = ch * 64;
        __syncthreads();   // previous chunk's P*V reads done before restaging
        const _Float16* Kg = K + ((size_t)head * Tpad + kb) * 64;
        const _Float16* Vg = V + ((size_t)head * Tpad + kb) * 64;
        for (int c2 = tid; c2 < 512; c2 += 128) {
            int r = c2 >> 3, co = (c2 & 7) * 8;
            *(v8h*)&Ks[r][co] = *(const v8h*)(Kg + (size_t)r * 64 + co);
            v8h vv = *(const v8h*)(Vg + (size_t)r * 64 + co);
#pragma unroll
            for (int e = 0; e < 8; ++e) Vt[co + e][r] = vv[e];
        }
        __syncthreads();

        // S = Q K^T (scaled, masked), kept in C fragments
        v8f sfrag[4];
#pragma unroll
        for (int nt = 0; nt < 4; ++nt) {
            v8f c = {};
#pragma unroll
            for (int kk = 0; kk < 2; ++kk) {
                v16h a = load_frag_a(&Qs[w * 16][kk * 32], 72, lane);
                v16h b = load_frag_b_t(&Ks[nt * 16][kk * 32], 72, lane);
                c = __builtin_amdgcn_wmma_f32_16x16x32_f16(
                    false, a, false, b, (short)0, c, false, false);
            }
            bool valid = (kb + nt * 16 + n) < Tvalid;
#pragma unroll
            for (int r = 0; r < 8; ++r) c[r] = valid ? c[r] * scale : -1e30f;
            sfrag[nt] = c;
        }

        // Online softmax in registers (row = mb + r for this lane's half)
        float mnew[8], alpha[8];
#pragma unroll
        for (int r = 0; r < 8; ++r) {
            float mx = fmaxf(fmaxf(sfrag[0][r], sfrag[1][r]),
                             fmaxf(sfrag[2][r], sfrag[3][r]));
            mx = fmaxf(mx, __shfl_xor(mx, 1, 32));
            mx = fmaxf(mx, __shfl_xor(mx, 2, 32));
            mx = fmaxf(mx, __shfl_xor(mx, 4, 32));
            mx = fmaxf(mx, __shfl_xor(mx, 8, 32));
            float mn = fmaxf(mrow[r], mx);
            mnew[r]  = mn;
            alpha[r] = __expf(mrow[r] - mn);   // first chunk: exp(-inf)=0
            mrow[r]  = mn;
        }
#pragma unroll
        for (int nt = 0; nt < 4; ++nt) {
            int jloc = nt * 16 + n;
#pragma unroll
            for (int r = 0; r < 8; ++r) {
                float pv = __expf(sfrag[nt][r] - mnew[r]);
                sfrag[nt][r] = pv;
                Psm[w][mb + r][jloc] = (_Float16)pv;
            }
        }
#pragma unroll
        for (int r = 0; r < 8; ++r) {
            float s = sfrag[0][r] + sfrag[1][r] + sfrag[2][r] + sfrag[3][r];
            s += __shfl_xor(s, 1, 32);
            s += __shfl_xor(s, 2, 32);
            s += __shfl_xor(s, 4, 32);
            s += __shfl_xor(s, 8, 32);
            lrow[r] = lrow[r] * alpha[r] + s;
            acc[0][r] *= alpha[r];
            acc[1][r] *= alpha[r];
            acc[2][r] *= alpha[r];
            acc[3][r] *= alpha[r];
        }
        __syncthreads();   // Psm visible across lanes before A-frag reads

        // O += P * V
#pragma unroll
        for (int nt = 0; nt < 4; ++nt) {
#pragma unroll
            for (int kk = 0; kk < 2; ++kk) {
                v16h a = load_frag_a(&Psm[w][0][kk * 32], 72, lane);
                v16h b = load_frag_b_t(&Vt[nt * 16][kk * 32], 72, lane);
                acc[nt] = __builtin_amdgcn_wmma_f32_16x16x32_f16(
                    false, a, false, b, (short)0, acc[nt], false, false);
            }
        }
    }

    float inv[8];
#pragma unroll
    for (int r = 0; r < 8; ++r) inv[r] = 1.f / lrow[r];
#pragma unroll
    for (int nt = 0; nt < 4; ++nt) {
        int d = nt * 16 + n;
        if (d < 40) {
#pragma unroll
            for (int r = 0; r < 8; ++r) {
                int gm = qbase + w * 16 + mb + r;
                O[(size_t)gm * 320 + head * 40 + d] = (_Float16)(acc[nt][r] * inv[r]);
            }
        }
    }
}

// ---------------------------------------------------------------------------
// Host orchestration
// ---------------------------------------------------------------------------
extern "C" void kernel_launch(void* const* d_in, const int* in_sizes, int n_in,
                              void* d_out, int out_size, void* d_ws, size_t ws_size,
                              hipStream_t stream) {
    (void)in_sizes; (void)n_in; (void)out_size; (void)ws_size;

    const float* x          = (const float*)d_in[0];
    const float* context    = (const float*)d_in[1];
    const float* gn_scale   = (const float*)d_in[2];
    const float* gn_bias    = (const float*)d_in[3];
    const float* w_proj_in  = (const float*)d_in[4];
    const float* b_proj_in  = (const float*)d_in[5];
    const float* ln1_s      = (const float*)d_in[6];
    const float* ln1_b      = (const float*)d_in[7];
    const float* wq1        = (const float*)d_in[8];
    const float* wk1        = (const float*)d_in[9];
    const float* wv1        = (const float*)d_in[10];
    const float* wo1        = (const float*)d_in[11];
    const float* bo1        = (const float*)d_in[12];
    const float* ln2_s      = (const float*)d_in[13];
    const float* ln2_b      = (const float*)d_in[14];
    const float* wq2        = (const float*)d_in[15];
    const float* wk2        = (const float*)d_in[16];
    const float* wv2        = (const float*)d_in[17];
    const float* wo2        = (const float*)d_in[18];
    const float* bo2        = (const float*)d_in[19];
    const float* ln3_s      = (const float*)d_in[20];
    const float* ln3_b      = (const float*)d_in[21];
    const float* w_ff1      = (const float*)d_in[22];
    const float* b_ff1      = (const float*)d_in[23];
    const float* w_ff2      = (const float*)d_in[24];
    const float* b_ff2      = (const float*)d_in[25];
    const float* w_proj_out = (const float*)d_in[26];
    const float* b_proj_out = (const float*)d_in[27];

    const int S = 4096, D = 320, FF = 2560, GG = 1280, CTXD = 768, CTXT = 77, TPAD = 128;
    const float SCALE = 0.15811388300841898f;  // 40^-0.5

    char* ws = (char*)d_ws;
    size_t off = 0;
    auto alloc = [&](size_t bytes) -> void* {
        void* p = ws + off;
        off = (off + bytes + 255) & ~(size_t)255;
        return p;
    };

    // f16 weight copies
    _Float16* w16_pi  = (_Float16*)alloc((size_t)D * D * 2);
    _Float16* w16_q1  = (_Float16*)alloc((size_t)D * D * 2);
    _Float16* w16_k1  = (_Float16*)alloc((size_t)D * D * 2);
    _Float16* w16_v1  = (_Float16*)alloc((size_t)D * D * 2);
    _Float16* w16_o1  = (_Float16*)alloc((size_t)D * D * 2);
    _Float16* w16_q2  = (_Float16*)alloc((size_t)D * D * 2);
    _Float16* w16_k2  = (_Float16*)alloc((size_t)CTXD * D * 2);
    _Float16* w16_v2  = (_Float16*)alloc((size_t)CTXD * D * 2);
    _Float16* w16_o2  = (_Float16*)alloc((size_t)D * D * 2);
    _Float16* w16_f1  = (_Float16*)alloc((size_t)D * FF * 2);
    _Float16* w16_f2  = (_Float16*)alloc((size_t)GG * D * 2);
    _Float16* w16_po  = (_Float16*)alloc((size_t)D * D * 2);
    // activations
    _Float16* gn16    = (_Float16*)alloc((size_t)S * D * 2);
    float*    hbuf    = (float*)   alloc((size_t)S * D * 4);
    _Float16* ln16    = (_Float16*)alloc((size_t)S * D * 2);
    _Float16* q16     = (_Float16*)alloc((size_t)8 * S * 64 * 2);
    _Float16* k16     = (_Float16*)alloc((size_t)8 * S * 64 * 2);
    _Float16* v16     = (_Float16*)alloc((size_t)8 * S * 64 * 2);
    _Float16* o16     = (_Float16*)alloc((size_t)S * D * 2);
    _Float16* ctx16   = (_Float16*)alloc((size_t)CTXT * CTXD * 2);
    _Float16* ffy16   = (_Float16*)alloc((size_t)S * FF * 2);
    _Float16* gg16    = (_Float16*)alloc((size_t)S * GG * 2);
    _Float16* h16     = (_Float16*)alloc((size_t)S * D * 2);

    auto conv = [&](const float* src, _Float16* dst, int nElem) {
        k_f32_to_f16<<<(nElem + 255) / 256, 256, 0, stream>>>(src, dst, nElem);
    };
    auto fill0 = [&](_Float16* dst, int nElem) {
        k_fill0_f16<<<(nElem + 255) / 256, 256, 0, stream>>>(dst, nElem);
    };
    auto gemm = [&](const _Float16* A, int lda, const _Float16* B, int ldb,
                    int M, int N, int Kd, const float* bias, const float* res,
                    float* outf, _Float16* outh, int ldc, int mode, int qkvms) {
        GemmP p{A, B, bias, res, outf, outh, M, N, Kd, lda, ldb, ldc, mode, qkvms};
        dim3 grid((N + 63) / 64, (M + 63) / 64);
        k_wmma_gemm<<<grid, 128, 0, stream>>>(p);
    };

    // Weight conversions (fp32 -> f16)
    conv(w_proj_in,  w16_pi, D * D);
    conv(wq1, w16_q1, D * D);  conv(wk1, w16_k1, D * D);
    conv(wv1, w16_v1, D * D);  conv(wo1, w16_o1, D * D);
    conv(wq2, w16_q2, D * D);
    conv(wk2, w16_k2, CTXD * D);  conv(wv2, w16_v2, CTXD * D);
    conv(wo2, w16_o2, D * D);
    conv(w_ff1, w16_f1, D * FF);  conv(w_ff2, w16_f2, GG * D);
    conv(w_proj_out, w16_po, D * D);
    conv(context, ctx16, CTXT * CTXD);

    // GroupNorm -> [s,c] f16
    k_groupnorm<<<32, 256, 0, stream>>>(x, gn_scale, gn_bias, gn16);

    // proj_in: h = GN(x)^T @ W_pi + b  (f32 residual stream)
    gemm(gn16, D, w16_pi, D, S, D, D, b_proj_in, nullptr, hbuf, nullptr, D, 0, 0);

    // ---- self-attention ----
    k_layernorm<<<S, 128, 0, stream>>>(hbuf, ln1_s, ln1_b, ln16);
    fill0(q16, 8 * S * 64); fill0(k16, 8 * S * 64); fill0(v16, 8 * S * 64);
    gemm(ln16, D, w16_q1, D, S, D, D, nullptr, nullptr, nullptr, q16, 0, 2, S);
    gemm(ln16, D, w16_k1, D, S, D, D, nullptr, nullptr, nullptr, k16, 0, 2, S);
    gemm(ln16, D, w16_v1, D, S, D, D, nullptr, nullptr, nullptr, v16, 0, 2, S);
    k_flash_attn<<<dim3(S / 64, 8), 128, 0, stream>>>(q16, k16, v16, o16, S, S, S, SCALE);
    gemm(o16, D, w16_o1, D, S, D, D, bo1, hbuf, hbuf, nullptr, D, 0, 0);

    // ---- cross-attention ----
    k_layernorm<<<S, 128, 0, stream>>>(hbuf, ln2_s, ln2_b, ln16);
    gemm(ln16, D, w16_q2, D, S, D, D, nullptr, nullptr, nullptr, q16, 0, 2, S);
    fill0(k16, 8 * TPAD * 64); fill0(v16, 8 * TPAD * 64);
    gemm(ctx16, CTXD, w16_k2, D, CTXT, D, CTXD, nullptr, nullptr, nullptr, k16, 0, 2, TPAD);
    gemm(ctx16, CTXD, w16_v2, D, CTXT, D, CTXD, nullptr, nullptr, nullptr, v16, 0, 2, TPAD);
    k_flash_attn<<<dim3(S / 64, 8), 128, 0, stream>>>(q16, k16, v16, o16, S, CTXT, TPAD, SCALE);
    gemm(o16, D, w16_o2, D, S, D, D, bo2, hbuf, hbuf, nullptr, D, 0, 0);

    // ---- GEGLU feed-forward ----
    k_layernorm<<<S, 128, 0, stream>>>(hbuf, ln3_s, ln3_b, ln16);
    gemm(ln16, D, w16_f1, FF, S, FF, D, b_ff1, nullptr, nullptr, ffy16, FF, 1, 0);
    k_geglu<<<(S * GG + 255) / 256, 256, 0, stream>>>(ffy16, gg16, S);
    gemm(gg16, GG, w16_f2, D, S, D, GG, b_ff2, hbuf, hbuf, nullptr, D, 0, 0);

    // ---- proj_out (transposed store back to [c,h,w] + input residual) ----
    conv(hbuf, h16, S * D);
    gemm(h16, D, w16_po, D, S, D, D, b_proj_out, x, (float*)d_out, nullptr, S, 3, 0);
}
